// MastDecoderLayer_3831110828360
// MI455X (gfx1250) — matrix operand order repaired and checked by reading, not compiled, not run on hardware
//
#include <hip/hip_runtime.h>
#include <hip/hip_bf16.h>

// ---------------------------------------------------------------------------
// MAST decoder layer for MI455X (gfx1250, wave32, WMMA).
// GEMMs: v_wmma_f32_16x16x32_f16 with f32 accumulate; weights pre-transposed
// to f16 [K][N] so B tiles stream into LDS via GLOBAL_LOAD_ASYNC_TO_LDS_B128
// (ASYNCcnt). All scratch in d_ws, all work on stream.
// ---------------------------------------------------------------------------

#define B_   16
#define ND_  300
#define NT_  64
#define C_   256
#define H_   8
#define HD_  32
#define L_   4
#define P_   4
#define DFF_ 1024
#define S_   12240   // 96*96 + 48*48 + 24*24 + 12*12

typedef __attribute__((ext_vector_type(16))) _Float16 v16h;
typedef __attribute__((ext_vector_type(8)))  _Float16 h8;
typedef __attribute__((ext_vector_type(8)))  float    v8f;
typedef __attribute__((ext_vector_type(4)))  int      v4i;

#if defined(__HIP_DEVICE_COMPILE__) && __has_builtin(__builtin_amdgcn_global_load_async_to_lds_b128)
#define USE_ASYNC_LDS 1
typedef __attribute__((address_space(1))) v4i* g_v4i_p;   // global
typedef __attribute__((address_space(3))) v4i* l_v4i_p;   // LDS
#else
#define USE_ASYNC_LDS 0
#endif

#if USE_ASYNC_LDS
__device__ __forceinline__ void wait_async0() {
#if __has_builtin(__builtin_amdgcn_s_wait_asynccnt)
  __builtin_amdgcn_s_wait_asynccnt(0);
#else
  asm volatile("s_wait_asynccnt 0x0" ::: "memory");
#endif
}
#endif

__device__ __forceinline__ float wave_reduce_sum(float v) {
  #pragma unroll
  for (int off = 16; off > 0; off >>= 1) v += __shfl_xor(v, off, 32);
  return v;
}

// ---------------------------------------------------------------------------
// Weight pre-transpose: src (N,K) f32 row-major  ->  dst (K,N) f16 row-major.
// ---------------------------------------------------------------------------
__global__ void transpose_w_kernel(const float* __restrict__ src,
                                   _Float16* __restrict__ dst, int N, int K) {
  const int i = blockIdx.x * 256 + threadIdx.x;
  if (i >= N * K) return;
  const int k = i / N;
  const int n = i % N;
  dst[i] = (_Float16)src[(size_t)n * K + k];
}

// ---------------------------------------------------------------------------
// WMMA GEMM:  Out[M,N] = A[M,K] * Wt[K, ncol0:ncol0+N] + bias[N]
//   A : f32 (M,K) row-major          Wt : f16 (K, ldw) row-major (pre-transposed)
// Block tile 64x128, 128 threads / 4 waves; wave owns 16 rows x 128 cols.
// Requires M%64==0, N%128==0, K%32==0 (true for every call here).
// ---------------------------------------------------------------------------
#define TM 64
#define TN 128
#define TK 32

__global__ __launch_bounds__(128) void wmma_gemm_kernel(
    const float* __restrict__ A, const _Float16* __restrict__ Wt, int ldw, int ncol0,
    const float* __restrict__ bias, float* __restrict__ Out,
    int M, int N, int K, int relu, const unsigned char* __restrict__ mask)
{
  __shared__ _Float16 As[TM][TK + 16];   // row stride 48 halves = 96B
  __shared__ _Float16 Bs[TK][TN + 8];    // [k][n], row stride 136 halves = 272B

  const int tid  = threadIdx.x;
  const int lane = tid & 31;
  const int wave = tid >> 5;
  const int bm = blockIdx.x * TM;
  const int bn = blockIdx.y * TN;

  v8f acc[TN / 16] = {};

  const int lr = tid >> 3;        // 0..15 : row within a 16-row slab
  const int lc = (tid & 7) * 4;   // 0,4,...,28 : K column base (float4)

  for (int k0 = 0; k0 < K; k0 += TK) {
    // ---- Stage A tile (TM x TK): f32 -> f16, packed 8-byte DS stores ----
    #pragma unroll
    for (int r = 0; r < TM; r += 16) {
      const float4 f = *(const float4*)(A + (size_t)(bm + r + lr) * K + k0 + lc);
      union { _Float16 h[4]; uint2 u; } pk;
      pk.h[0] = (_Float16)f.x; pk.h[1] = (_Float16)f.y;
      pk.h[2] = (_Float16)f.z; pk.h[3] = (_Float16)f.w;
      *(uint2*)&As[r + lr][lc] = pk.u;
    }
    if (k0 + TK < K) {  // prefetch next A k-slice (global_prefetch_b8)
      __builtin_prefetch(A + (size_t)(bm + lr) * K + k0 + TK + lc, 0, 1);
    }
    // ---- Stage B tile (TK x TN) from pre-transposed f16: raw 16B chunks ----
    #pragma unroll
    for (int i = 0; i < (TK * TN) / (8 * 128); ++i) {   // 4 chunks / thread
      const int ch = tid + 128 * i;
      const int kr = ch >> 4;           // 0..31
      const int c8 = (ch & 15) * 8;     // half offset 0..120
      const _Float16* gsrc = Wt + (size_t)(k0 + kr) * ldw + ncol0 + bn + c8;
#if USE_ASYNC_LDS
      __builtin_amdgcn_global_load_async_to_lds_b128(
          (g_v4i_p)(v4i*)gsrc, (l_v4i_p)(v4i*)&Bs[kr][c8], 0, 0);
#else
      *(h8*)&Bs[kr][c8] = *(const h8*)gsrc;
#endif
    }
#if USE_ASYNC_LDS
    wait_async0();
#endif
    __syncthreads();

    // ---- A fragment (16x32): lane m=lane&15; K chunks half*8 and 16+half*8 ----
    const int m16   = lane & 15;
    const int half8 = (lane >> 4) * 8;
    union { v16h v; h8 h[2]; } au;
    au.h[0] = *(const h8*)&As[wave * 16 + m16][half8];
    au.h[1] = *(const h8*)&As[wave * 16 + m16][16 + half8];
    const v16h af = au.v;

    // ---- Preload all 8 B fragments, then issue 8 back-to-back WMMAs ----
    v16h bfr[TN / 16];
    #pragma unroll
    for (int j = 0; j < TN / 16; ++j) {
      v16h bv;
      #pragma unroll
      for (int s = 0; s < 8; ++s) {
        bv[s]     = Bs[half8 + s][j * 16 + m16];
        bv[8 + s] = Bs[16 + half8 + s][j * 16 + m16];
      }
      bfr[j] = bv;
    }
    #pragma unroll
    for (int j = 0; j < TN / 16; ++j) {
      acc[j] = __builtin_amdgcn_wmma_f32_16x16x32_f16(
          false, af, false, bfr[j], (short)0, acc[j], false, false);
    }
    __syncthreads();
  }

  // ---- Epilogue. D layout: VGPR r, lane l -> M = r + 8*(l>=16), N = l&15 ----
  const int mbase = bm + wave * 16 + 8 * (lane >> 4);
  float keep[8];
  #pragma unroll
  for (int r = 0; r < 8; ++r)
    keep[r] = (mask && mask[mbase + r]) ? 0.0f : 1.0f;
  #pragma unroll
  for (int j = 0; j < TN / 16; ++j) {
    const int ncol = bn + j * 16 + (lane & 15);
    const float bv = bias[ncol];
    #pragma unroll
    for (int r = 0; r < 8; ++r) {
      float v = (acc[j][r] + bv) * keep[r];
      if (relu) v = fmaxf(v, 0.0f);
      Out[(size_t)(mbase + r) * N + ncol] = v;
    }
  }
}

// ---------------------------------------------------------------------------
// Multi-head attention: one thread per query row, online softmax, HD=32.
// ---------------------------------------------------------------------------
__global__ __launch_bounds__(128) void attn_kernel(
    const float* __restrict__ Q, const float* __restrict__ K,
    const float* __restrict__ V, float* __restrict__ O, int Lq, int Lk)
{
  const int bh = blockIdx.x;
  const int b  = bh / H_;
  const int h  = bh % H_;
  const int qi = blockIdx.y * 128 + threadIdx.x;
  if (qi >= Lq) return;

  const float scale = 0.17677669529663687f;  // 32^-0.5
  float qr[HD_];
  const float* qp = Q + ((size_t)(b * Lq + qi)) * C_ + h * HD_;
  #pragma unroll
  for (int d = 0; d < HD_; ++d) qr[d] = qp[d] * scale;

  float m = -3.0e38f, l = 0.0f;
  float acc[HD_];
  #pragma unroll
  for (int d = 0; d < HD_; ++d) acc[d] = 0.0f;

  for (int j = 0; j < Lk; ++j) {
    const float* kp = K + ((size_t)(b * Lk + j)) * C_ + h * HD_;
    float s = 0.0f;
    #pragma unroll
    for (int d = 0; d < HD_; ++d) s += qr[d] * kp[d];
    const float mn   = fmaxf(m, s);
    const float corr = __expf(m - mn);
    const float p    = __expf(s - mn);
    l = l * corr + p;
    const float* vp = V + ((size_t)(b * Lk + j)) * C_ + h * HD_;
    #pragma unroll
    for (int d = 0; d < HD_; ++d) acc[d] = acc[d] * corr + p * vp[d];
    m = mn;
  }
  const float inv = 1.0f / l;
  float* op = O + ((size_t)(b * Lq + qi)) * C_ + h * HD_;
  #pragma unroll
  for (int d = 0; d < HD_; ++d) op[d] = acc[d] * inv;
}

// ---------------------------------------------------------------------------
// LayerNorm over C=256, one wave32 per row, optional residual.
// ---------------------------------------------------------------------------
__global__ __launch_bounds__(256) void ln_kernel(
    float* __restrict__ out, const float* __restrict__ x,
    const float* __restrict__ res, const float* __restrict__ w,
    const float* __restrict__ b, int rows)
{
  const int lane = threadIdx.x & 31;
  const int row  = blockIdx.x * 8 + (threadIdx.x >> 5);
  if (row >= rows) return;
  const size_t base = (size_t)row * C_;

  float vals[8];
  float s = 0.0f;
  #pragma unroll
  for (int i = 0; i < 8; ++i) {
    const int c = lane + i * 32;
    float v = x[base + c];
    if (res) v += res[base + c];
    vals[i] = v; s += v;
  }
  s = wave_reduce_sum(s);
  const float mean = s * (1.0f / C_);
  float var = 0.0f;
  #pragma unroll
  for (int i = 0; i < 8; ++i) { const float d = vals[i] - mean; var += d * d; }
  var = wave_reduce_sum(var) * (1.0f / C_);
  const float rstd = rsqrtf(var + 1e-5f);
  #pragma unroll
  for (int i = 0; i < 8; ++i) {
    const int c = lane + i * 32;
    out[base + c] = (vals[i] - mean) * rstd * w[c] + b[c];
  }
}

// ---------------------------------------------------------------------------
// Elementwise add and sequence-dim concat
// ---------------------------------------------------------------------------
__global__ void add_kernel(float* __restrict__ out, const float* __restrict__ a,
                           const float* __restrict__ b, int n) {
  const int i = blockIdx.x * 256 + threadIdx.x;
  if (i < n) out[i] = a[i] + b[i];
}

__global__ void concat_kernel(float* __restrict__ out, const float* __restrict__ a,
                              const float* __restrict__ bsrc, int La, int Lb) {
  const int Lt = La + Lb;
  const int i = blockIdx.x * 256 + threadIdx.x;
  if (i >= B_ * Lt * C_) return;
  const int c = i % C_;
  const int r = (i / C_) % Lt;
  const int b = i / (C_ * Lt);
  out[i] = (r < La) ? a[((size_t)(b * La + r)) * C_ + c]
                    : bsrc[((size_t)(b * Lb + (r - La))) * C_ + c];
}

// ---------------------------------------------------------------------------
// MSDA sampling: one wave32 per (b, query, head); lane = channel d (HD=32).
// ---------------------------------------------------------------------------
__global__ __launch_bounds__(128) void msda_kernel(
    const float* __restrict__ value, const float* __restrict__ off,
    const float* __restrict__ awraw, const float* __restrict__ refp,
    float* __restrict__ out)
{
  const int gw   = (blockIdx.x * 128 + threadIdx.x) >> 5;
  const int lane = threadIdx.x & 31;
  const int h  = gw % H_;
  const int t  = gw / H_;
  const int qi = t % ND_;
  const int b  = t / ND_;
  if (b >= B_) return;

  const int SH[4] = {96, 48, 24, 12};
  const int ST[4] = {0, 9216, 11520, 12096};

  const float* awp = awraw + ((size_t)(b * ND_ + qi)) * (H_ * 16) + h * 16;
  float aw[16];
  float mx = -3.0e38f;
  #pragma unroll
  for (int i = 0; i < 16; ++i) { aw[i] = awp[i]; mx = fmaxf(mx, aw[i]); }
  float sum = 0.0f;
  #pragma unroll
  for (int i = 0; i < 16; ++i) { aw[i] = __expf(aw[i] - mx); sum += aw[i]; }
  const float inv = 1.0f / sum;

  const float* offp = off + ((size_t)(b * ND_ + qi)) * (H_ * 32) + h * 32;
  float acc = 0.0f;
  #pragma unroll
  for (int l = 0; l < L_; ++l) {
    const int Wl = SH[l];
    const int Hl = SH[l];
    const float rx = refp[(((size_t)(b * ND_ + qi)) * L_ + l) * 2 + 0];
    const float ry = refp[(((size_t)(b * ND_ + qi)) * L_ + l) * 2 + 1];
    const float* vb = value + ((size_t)(b * S_ + ST[l])) * C_ + h * HD_ + lane;
    #pragma unroll
    for (int p = 0; p < P_; ++p) {
      const float ox = offp[(l * P_ + p) * 2 + 0];
      const float oy = offp[(l * P_ + p) * 2 + 1];
      const float px = (rx + ox / (float)Wl) * (float)Wl - 0.5f;
      const float py = (ry + oy / (float)Hl) * (float)Hl - 0.5f;
      const float x0f = floorf(px), y0f = floorf(py);
      const float fx = px - x0f, fy = py - y0f;
      const int x0 = (int)x0f, y0 = (int)y0f;
      float samp = 0.0f;
      #pragma unroll
      for (int dy = 0; dy < 2; ++dy) {
        #pragma unroll
        for (int dx = 0; dx < 2; ++dx) {
          const int xi = x0 + dx, yi = y0 + dy;
          if (xi >= 0 && xi < Wl && yi >= 0 && yi < Hl) {
            const float wgt = (dx ? fx : 1.0f - fx) * (dy ? fy : 1.0f - fy);
            samp += wgt * vb[(size_t)(yi * Wl + xi) * C_];
          }
        }
      }
      acc += aw[l * P_ + p] * inv * samp;
    }
  }
  out[((size_t)(b * ND_ + qi)) * C_ + h * HD_ + lane] = acc;
}

// ---------------------------------------------------------------------------
// Host orchestration
// ---------------------------------------------------------------------------
extern "C" void kernel_launch(void* const* d_in, const int* in_sizes, int n_in,
                              void* d_out, int out_size, void* d_ws, size_t ws_size,
                              hipStream_t stream) {
  (void)in_sizes; (void)n_in; (void)out_size; (void)ws_size;

  const float* DQ   = (const float*)d_in[0];
  const float* TQ   = (const float*)d_in[1];
  const float* DP   = (const float*)d_in[2];
  const float* TS   = (const float*)d_in[3];
  const float* RP   = (const float*)d_in[4];
  const float* SRC  = (const float*)d_in[5];
  const unsigned char* MASK = (const unsigned char*)d_in[8];

  const float* d_sa_w  = (const float*)d_in[9];
  const float* d_sa_b  = (const float*)d_in[10];
  const float* d_sa_ow = (const float*)d_in[11];
  const float* d_sa_ob = (const float*)d_in[12];
  const float* t_sa_w  = (const float*)d_in[13];
  const float* t_sa_b  = (const float*)d_in[14];
  const float* t_sa_ow = (const float*)d_in[15];
  const float* t_sa_ob = (const float*)d_in[16];
  const float* off_w   = (const float*)d_in[17];
  const float* off_bb  = (const float*)d_in[18];
  const float* aw_w    = (const float*)d_in[19];
  const float* aw_bb   = (const float*)d_in[20];
  const float* val_w   = (const float*)d_in[21];
  const float* val_b   = (const float*)d_in[22];
  const float* out_w   = (const float*)d_in[23];
  const float* out_bb  = (const float*)d_in[24];
  const float* d_ffn_w1 = (const float*)d_in[25];
  const float* d_ffn_b1 = (const float*)d_in[26];
  const float* d_ffn_w2 = (const float*)d_in[27];
  const float* d_ffn_b2 = (const float*)d_in[28];
  const float* t_ffn_w1 = (const float*)d_in[29];
  const float* t_ffn_b1 = (const float*)d_in[30];
  const float* t_ffn_w2 = (const float*)d_in[31];
  const float* t_ffn_b2 = (const float*)d_in[32];
  const float* ln_dsa_w  = (const float*)d_in[33];
  const float* ln_dsa_b  = (const float*)d_in[34];
  const float* ln_tsa_w  = (const float*)d_in[35];
  const float* ln_tsa_b  = (const float*)d_in[36];
  const float* ln_dca_w  = (const float*)d_in[37];
  const float* ln_dca_b  = (const float*)d_in[38];
  const float* ln_tskip_w = (const float*)d_in[39];
  const float* ln_tskip_b = (const float*)d_in[40];
  const float* ln_dffn_w = (const float*)d_in[41];
  const float* ln_dffn_b = (const float*)d_in[42];
  const float* ln_tffn_w = (const float*)d_in[43];
  const float* ln_tffn_b = (const float*)d_in[44];

  const int NE_D = B_ * ND_ * C_;        // 1,228,800
  const int NE_T = B_ * NT_ * C_;        // 262,144
  const int LA   = ND_ + NT_;            // 364
  const int NE_A = B_ * LA * C_;         // 1,490,944

  // ---- workspace bump allocator (with planned reuse) ----
  float* wsf = (float*)d_ws;
  size_t wo = 0;
  auto alloc = [&](size_t n) { float* p = wsf + wo; wo += n; return p; };
  float* q1    = alloc(NE_D);            // detect q+pos; later dq after cross-LN
  float* qh_d  = alloc(NE_D);            // later MSDA offsets
  float* kh_d  = alloc(NE_D);            // later MSDA raw attn weights
  float* vh_d  = alloc(NE_D);            // later detect FFN output
  float* attnO = alloc(NE_D);            // later MSDA sampled output
  float* proj  = alloc(NE_D);            // later cross-attn projection
  float* dq    = alloc(NE_D);
  float* q2    = alloc(NE_D);
  float* tqq   = alloc(NE_T);            // later tq after skip-LN
  float* allk  = alloc(NE_A);            // later track FFN hidden
  float* allv  = alloc(NE_A);            // later track FFN output
  float* qh_t  = alloc(NE_T);
  float* kh_t  = alloc(NE_A);
  float* vh_t  = alloc(NE_A);
  float* attnT = alloc(NE_T);
  float* tproj = alloc(NE_T);
  float* tq    = alloc(NE_T);
  float* ffnh_d = alloc((size_t)B_ * ND_ * DFF_);
  // pre-transposed f16 weights (counts are halves; alloc in floats)
  _Float16* wt_dsa   = (_Float16*)alloc(768 * 256 / 2);
  _Float16* wt_dsaow = (_Float16*)alloc(256 * 256 / 2);
  _Float16* wt_tsa   = (_Float16*)alloc(768 * 256 / 2);
  _Float16* wt_tsaow = (_Float16*)alloc(256 * 256 / 2);
  _Float16* wt_off   = (_Float16*)alloc(256 * 256 / 2);
  _Float16* wt_aw    = (_Float16*)alloc(128 * 256 / 2);
  _Float16* wt_val   = (_Float16*)alloc(256 * 256 / 2);
  _Float16* wt_out   = (_Float16*)alloc(256 * 256 / 2);
  _Float16* wt_dff1  = (_Float16*)alloc(1024 * 256 / 2);
  _Float16* wt_dff2  = (_Float16*)alloc(256 * 1024 / 2);
  _Float16* wt_tff1  = (_Float16*)alloc(1024 * 256 / 2);
  _Float16* wt_tff2  = (_Float16*)alloc(256 * 1024 / 2);
  float* value = alloc((size_t)B_ * S_ * C_);   // 200 MB, the big one

  auto tw = [&](const float* src, _Float16* dst, int N, int K) {
    transpose_w_kernel<<<dim3((N * K + 255) / 256), dim3(256), 0, stream>>>(src, dst, N, K);
  };
  auto gemm = [&](const float* A, const _Float16* Wt, int ldw, int ncol0,
                  const float* bias, float* Out, int M, int N, int K,
                  int relu, const unsigned char* mask) {
    dim3 grid(M / TM, N / TN);
    wmma_gemm_kernel<<<grid, dim3(128), 0, stream>>>(A, Wt, ldw, ncol0, bias, Out,
                                                     M, N, K, relu, mask);
  };
  auto addv = [&](float* o, const float* a, const float* b, int n) {
    add_kernel<<<dim3((n + 255) / 256), dim3(256), 0, stream>>>(o, a, b, n);
  };
  auto ln = [&](float* o, const float* x, const float* r, const float* w,
                const float* b, int rows) {
    ln_kernel<<<dim3((rows + 7) / 8), dim3(256), 0, stream>>>(o, x, r, w, b, rows);
  };

  const int MD = B_ * ND_;   // 4800
  const int MT = B_ * NT_;   // 1024
  const int MA = B_ * LA;    // 5824
  const int MV = B_ * S_;    // 195840

  // ===== 0) Pre-transpose all GEMM weights to f16 [K][N] =====
  tw(d_sa_w,   wt_dsa,   768,  256);
  tw(d_sa_ow,  wt_dsaow, 256,  256);
  tw(t_sa_w,   wt_tsa,   768,  256);
  tw(t_sa_ow,  wt_tsaow, 256,  256);
  tw(off_w,    wt_off,   256,  256);
  tw(aw_w,     wt_aw,    128,  256);
  tw(val_w,    wt_val,   256,  256);
  tw(out_w,    wt_out,   256,  256);
  tw(d_ffn_w1, wt_dff1,  1024, 256);
  tw(d_ffn_w2, wt_dff2,  256,  1024);
  tw(t_ffn_w1, wt_tff1,  1024, 256);
  tw(t_ffn_w2, wt_tff2,  256,  1024);

  // ===== 1) Detect self-attention =====
  addv(q1, DQ, DP, NE_D);
  gemm(q1, wt_dsa, 768, 0,   d_sa_b,       qh_d, MD, C_, C_, 0, nullptr);
  gemm(q1, wt_dsa, 768, 256, d_sa_b + 256, kh_d, MD, C_, C_, 0, nullptr);
  gemm(DQ, wt_dsa, 768, 512, d_sa_b + 512, vh_d, MD, C_, C_, 0, nullptr);
  attn_kernel<<<dim3(B_ * H_, (ND_ + 127) / 128), dim3(128), 0, stream>>>(
      qh_d, kh_d, vh_d, attnO, ND_, ND_);
  gemm(attnO, wt_dsaow, 256, 0, d_sa_ob, proj, MD, C_, C_, 0, nullptr);
  ln(dq, DQ, proj, ln_dsa_w, ln_dsa_b, MD);

  // ===== 2) Track self-attention =====
  addv(tqq, TQ, TS, NE_T);
  addv(q2, dq, DP, NE_D);
  concat_kernel<<<dim3((NE_A + 255) / 256), dim3(256), 0, stream>>>(allk, q2, tqq, ND_, NT_);
  concat_kernel<<<dim3((NE_A + 255) / 256), dim3(256), 0, stream>>>(allv, dq, TQ, ND_, NT_);
  gemm(tqq,  wt_tsa, 768, 0,   t_sa_b,       qh_t, MT, C_, C_, 0, nullptr);
  gemm(allk, wt_tsa, 768, 256, t_sa_b + 256, kh_t, MA, C_, C_, 0, nullptr);
  gemm(allv, wt_tsa, 768, 512, t_sa_b + 512, vh_t, MA, C_, C_, 0, nullptr);
  attn_kernel<<<dim3(B_ * H_, 1), dim3(128), 0, stream>>>(qh_t, kh_t, vh_t, attnT, NT_, LA);
  gemm(attnT, wt_tsaow, 256, 0, t_sa_ob, tproj, MT, C_, C_, 0, nullptr);
  ln(tq, TQ, tproj, ln_tsa_w, ln_tsa_b, MT);

  // ===== 3) Multi-scale deformable cross-attention =====
  gemm(SRC, wt_val, 256, 0, val_b, value, MV, C_, C_, 0, MASK);
  float* offb = qh_d;
  float* awb  = kh_d;
  gemm(q2, wt_off, 256, 0, off_bb, offb, MD, C_,  C_, 0, nullptr);
  gemm(q2, wt_aw,  128, 0, aw_bb,  awb,  MD, 128, C_, 0, nullptr);
  float* msda = attnO;
  msda_kernel<<<dim3(B_ * ND_ * H_ / 4), dim3(128), 0, stream>>>(value, offb, awb, RP, msda);
  float* cross = proj;
  gemm(msda, wt_out, 256, 0, out_bb, cross, MD, C_, C_, 0, nullptr);
  float* dqA = q1;
  ln(dqA, dq, cross, ln_dca_w, ln_dca_b, MD);
  float* tq2 = tqq;
  ln(tq2, tq, nullptr, ln_tskip_w, ln_tskip_b, MT);

  // ===== 4) FFNs + final LayerNorms =====
  gemm(dqA, wt_dff1, 1024, 0, d_ffn_b1, ffnh_d, MD, DFF_, C_, 1, nullptr);
  float* ffo_d = vh_d;
  gemm(ffnh_d, wt_dff2, 256, 0, d_ffn_b2, ffo_d, MD, C_, DFF_, 0, nullptr);
  ln((float*)d_out, dqA, ffo_d, ln_dffn_w, ln_dffn_b, MD);

  float* ffnh_t = allk;
  gemm(tq2, wt_tff1, 1024, 0, t_ffn_b1, ffnh_t, MT, DFF_, C_, 1, nullptr);
  float* ffo_t = allv;
  gemm(ffnh_t, wt_tff2, 256, 0, t_ffn_b2, ffo_t, MT, C_, DFF_, 0, nullptr);
  ln((float*)d_out + NE_D, tq2, ffo_t, ln_tffn_w, ln_tffn_b, MT);
}
